// AttentionBlock_86328842649775
// MI455X (gfx1250) — matrix operand order repaired and checked by reading, not compiled, not run on hardware
//
#include <hip/hip_runtime.h>
#include <hip/hip_bf16.h>

// MI455X / gfx1250 attention block.
// Precision: f16 WMMA (v_wmma_f32_16x16x32_f16) with f32 accumulation; softmax/LN in f32.

typedef __attribute__((ext_vector_type(16))) _Float16 v16h;
typedef __attribute__((ext_vector_type(8)))  _Float16 v8h;
typedef __attribute__((ext_vector_type(8)))  float    v8f;

constexpr int S  = 2048;
constexpr int Dm = 1024;
constexpr int H  = 16;
constexpr int DH = 64;
constexpr int LROW = 2048 + 4;   // padded LDS row stride (floats) -> conflict-free frag gathers

__device__ __forceinline__ v8f wmma_f16(v16h a, v16h b, v8f c) {
    return __builtin_amdgcn_wmma_f32_16x16x32_f16(false, a, false, b, (short)0, c, false, false);
}

// A-fragment (16x32 f16, MxK): 16 contiguous K values p[0..15]
__device__ __forceinline__ v16h frag_ld_gap(const _Float16* p) {
    v8h a = *(const v8h*)(p);
    v8h b = *(const v8h*)(p + 8);
    return __builtin_shufflevector(a, b, 0,1,2,3,4,5,6,7,8,9,10,11,12,13,14,15);
}
// B-fragment (32x16 f16, KxN): 16 contiguous K values for this lane's column
__device__ __forceinline__ v16h frag_ld_contig(const _Float16* p) {
    v8h a = *(const v8h*)(p);
    v8h b = *(const v8h*)(p + 8);
    return __builtin_shufflevector(a, b, 0,1,2,3,4,5,6,7,8,9,10,11,12,13,14,15);
}

__global__ void cast_f16_kernel(const float* __restrict__ src, _Float16* __restrict__ dst,
                                int n, float scale) {
    int i = blockIdx.x * 256 + threadIdx.x;
    if (i < n) dst[i] = (_Float16)(src[i] * scale);
}

// C[M,N] = A[M,K] @ W[N,K]^T   (all f16 in, f32 accumulate)
// MODE 0: outH row-major [M,N] f16
// MODE 1: outH transposed [N,M] f16   (per-head-transposed V)
// MODE 2: outF row-major [M,N] f32, += resid
template <int MODE>
__global__ __launch_bounds__(128) void gemm_xWT(const _Float16* __restrict__ A,
                                                const _Float16* __restrict__ W,
                                                _Float16* __restrict__ outH,
                                                float* __restrict__ outF,
                                                const float* __restrict__ resid,
                                                int M, int N, int K) {
    const int lane = threadIdx.x & 31;
    const int wv   = threadIdx.x >> 5;
    const int ln   = lane & 15;
    const int hl   = lane >> 4;
    const int mbase = blockIdx.x * 16;
    const int nbase = blockIdx.y * 256 + wv * 64;   // wave tile: 16(M) x 64(N)

    v8f acc[4] = {v8f{}, v8f{}, v8f{}, v8f{}};
    const _Float16* arow = A + (size_t)(mbase + ln) * K + hl * 16;
    const _Float16* wb   = W + hl * 16;

    for (int kk = 0; kk < K; kk += 32) {
        v16h a = frag_ld_gap(arow + kk);
#pragma unroll
        for (int j = 0; j < 4; ++j) {
            v16h b = frag_ld_contig(wb + (size_t)(nbase + j * 16 + ln) * K + kk);
            acc[j] = wmma_f16(a, b, acc[j]);
        }
    }

#pragma unroll
    for (int r = 0; r < 8; ++r) {
        const int gm = mbase + r + 8 * hl;
#pragma unroll
        for (int j = 0; j < 4; ++j) {
            const int gn = nbase + j * 16 + ln;
            if (MODE == 0)      outH[(size_t)gm * N + gn] = (_Float16)acc[j][r];
            else if (MODE == 1) outH[(size_t)gn * M + gm] = (_Float16)acc[j][r];
            else { size_t idx = (size_t)gm * N + gn; outF[idx] = acc[j][r] + resid[idx]; }
        }
    }
}

// One block per (head, 16-row q tile). 4 waves. Full 16x2048 logit stripe in LDS.
__global__ __launch_bounds__(128) void attn_kernel(const _Float16* __restrict__ Q,   // [S,D] (pre-scaled)
                                                   const _Float16* __restrict__ Km,  // [S,D]
                                                   const _Float16* __restrict__ Vt,  // [D,S] = [h*DH+d][t]
                                                   const float* __restrict__ bias,   // [S,H*S]
                                                   const unsigned char* __restrict__ mask, // [S,S] bool
                                                   _Float16* __restrict__ attnO) {   // [S,D]
    extern __shared__ float smem[];
    float* slog = smem;                 // [16][LROW]
    float* srow = smem + 16 * LROW;     // [16] row sums

    const int tid  = threadIdx.x;
    const int lane = tid & 31;
    const int wv   = tid >> 5;
    const int ln   = lane & 15;
    const int hl   = lane >> 4;
    const int sbase = blockIdx.x * 16;
    const int h     = blockIdx.y;
    const int h64   = h * DH;

    // Q fragments (K = DH = 64 -> two k-steps of 32), loaded once
    const _Float16* qrow = Q + (size_t)(sbase + ln) * Dm + h64 + hl * 16;
    const v16h qa0 = frag_ld_gap(qrow);
    const v16h qa1 = frag_ld_gap(qrow + 32);

    // ---- Phase 1: logits = Q K^T + bias, mask, -> LDS. Each wave owns 512 t-columns.
    for (int tt = 0; tt < 32; ++tt) {
        const int tbase = wv * 512 + tt * 16;
        const _Float16* krow = Km + (size_t)(tbase + ln) * Dm + h64 + hl * 16;
        v8f acc = {};
        acc = wmma_f16(qa0, frag_ld_contig(krow),      acc);
        acc = wmma_f16(qa1, frag_ld_contig(krow + 32), acc);

        // prefetch next tile's bias stream (the 256MB HBM stream is the roofline floor)
        if (tt + 1 < 32)
            __builtin_prefetch(bias + (size_t)(sbase + 8 * hl) * (H * S) + (size_t)h * S + tbase + 16 + ln, 0, 1);

#pragma unroll
        for (int r = 0; r < 8; ++r) {
            const int m  = r + 8 * hl;
            const int gm = sbase + m;
            const int gn = tbase + ln;
            float v = acc[r] + bias[(size_t)gm * (H * S) + (size_t)h * S + gn];
            v = mask[(size_t)gm * S + gn] ? v : -3.4028235e38f;
            slog[m * LROW + gn] = v;
        }
    }
    __syncthreads();

    // ---- Phase 2: softmax per row (unnormalized exp stored back; sum kept in srow)
    {
        const int row = tid >> 3, sub = tid & 7;   // 8 threads per row, same wave
        float* rptr = slog + row * LROW;
        float rmax = -3.4028235e38f;
        for (int t = sub; t < S; t += 8) rmax = fmaxf(rmax, rptr[t]);
#pragma unroll
        for (int o = 1; o < 8; o <<= 1) rmax = fmaxf(rmax, __shfl_xor(rmax, o, 8));
        float rsum = 0.f;
        for (int t = sub; t < S; t += 8) {
            float e = __expf(rptr[t] - rmax);
            rptr[t] = e;
            rsum += e;
        }
#pragma unroll
        for (int o = 1; o < 8; o <<= 1) rsum += __shfl_xor(rsum, o, 8);
        if (sub == 0) srow[row] = rsum;
    }
    __syncthreads();

    // ---- Phase 3: O = weights @ V. Each wave owns 16 of the 64 head dims.
    const int dbase = wv * 16;
    const _Float16* vrow = Vt + (size_t)(h64 + dbase + ln) * S + hl * 16;
    const float* wrow = slog + ln * LROW + hl * 16;
    v8f o = {};
    for (int tb = 0; tb < S; tb += 32) {
        const float* p = wrow + tb;
        v16h wa;
#pragma unroll
        for (int i = 0; i < 8; ++i) { wa[i] = (_Float16)p[i]; wa[i + 8] = (_Float16)p[8 + i]; }
        o = wmma_f16(wa, frag_ld_contig(vrow + tb), o);
    }
#pragma unroll
    for (int r = 0; r < 8; ++r) {
        const int m = r + 8 * hl;
        const float inv = 1.0f / srow[m];
        attnO[(size_t)(sbase + m) * Dm + h64 + dbase + ln] = (_Float16)(o[r] * inv);
    }
}

// Per-row LayerNorm: one block per row of 1024.
__global__ __launch_bounds__(256) void ln_kernel(const float* __restrict__ res,
                                                 const float* __restrict__ w,
                                                 const float* __restrict__ b,
                                                 float* __restrict__ out) {
    __shared__ float red[256];
    const int row = blockIdx.x, tid = threadIdx.x;
    const float* r = res + (size_t)row * Dm;
    float v[4];
#pragma unroll
    for (int i = 0; i < 4; ++i) v[i] = r[tid + 256 * i];

    red[tid] = v[0] + v[1] + v[2] + v[3];
    __syncthreads();
    for (int st = 128; st > 0; st >>= 1) { if (tid < st) red[tid] += red[tid + st]; __syncthreads(); }
    const float mean = red[0] * (1.0f / Dm);
    __syncthreads();

    float s2 = 0.f;
#pragma unroll
    for (int i = 0; i < 4; ++i) { float d = v[i] - mean; s2 += d * d; }
    red[tid] = s2;
    __syncthreads();
    for (int st = 128; st > 0; st >>= 1) { if (tid < st) red[tid] += red[tid + st]; __syncthreads(); }
    const float rstd = rsqrtf(red[0] * (1.0f / Dm) + 1e-5f);

#pragma unroll
    for (int i = 0; i < 4; ++i) {
        const int c = tid + 256 * i;
        out[(size_t)row * Dm + c] = (v[i] - mean) * rstd * w[c] + b[c];
    }
}

extern "C" void kernel_launch(void* const* d_in, const int* in_sizes, int n_in,
                              void* d_out, int out_size, void* d_ws, size_t ws_size,
                              hipStream_t stream) {
    (void)in_sizes; (void)n_in; (void)out_size; (void)ws_size;
    const float* x    = (const float*)d_in[0];
    const float* bias = (const float*)d_in[1];
    const unsigned char* mask = (const unsigned char*)d_in[2]; // numpy bool = 1 byte
    const float* Wq = (const float*)d_in[3];
    const float* Wk = (const float*)d_in[4];
    const float* Wv = (const float*)d_in[5];
    const float* Wo = (const float*)d_in[6];
    const float* lnw = (const float*)d_in[7];
    const float* lnb = (const float*)d_in[8];
    float* out = (float*)d_out;

    // workspace layout (f16 tensors, 256B aligned by construction)
    char* ws = (char*)d_ws;
    _Float16* x16  = (_Float16*)(ws);                         // S*D
    _Float16* Wq16 = (_Float16*)(ws + (4u << 20));            // D*D
    _Float16* Wk16 = (_Float16*)(ws + (6u << 20));
    _Float16* Wv16 = (_Float16*)(ws + (8u << 20));
    _Float16* Wo16 = (_Float16*)(ws + (10u << 20));
    _Float16* Q16  = (_Float16*)(ws + (12u << 20));           // S*D (pre-scaled by 1/sqrt(DH))
    _Float16* K16  = (_Float16*)(ws + (16u << 20));
    _Float16* Vt16 = (_Float16*)(ws + (20u << 20));           // [D][S] transposed
    _Float16* at16 = (_Float16*)(ws + (24u << 20));           // attn [S,D]
    float*    res  = (float*)   (ws + (28u << 20));           // residual f32 [S,D]

    const int nXD = S * Dm, nDD = Dm * Dm;
    cast_f16_kernel<<<(nXD + 255) / 256, 256, 0, stream>>>(x,  x16,  nXD, 1.0f);
    cast_f16_kernel<<<(nDD + 255) / 256, 256, 0, stream>>>(Wq, Wq16, nDD, 0.125f); // fold 1/sqrt(64)
    cast_f16_kernel<<<(nDD + 255) / 256, 256, 0, stream>>>(Wk, Wk16, nDD, 1.0f);
    cast_f16_kernel<<<(nDD + 255) / 256, 256, 0, stream>>>(Wv, Wv16, nDD, 1.0f);
    cast_f16_kernel<<<(nDD + 255) / 256, 256, 0, stream>>>(Wo, Wo16, nDD, 1.0f);

    dim3 gGemm(S / 16, Dm / 256);
    gemm_xWT<0><<<gGemm, 128, 0, stream>>>(x16, Wq16, Q16,  nullptr, nullptr, S, Dm, Dm);
    gemm_xWT<0><<<gGemm, 128, 0, stream>>>(x16, Wk16, K16,  nullptr, nullptr, S, Dm, Dm);
    gemm_xWT<1><<<gGemm, 128, 0, stream>>>(x16, Wv16, Vt16, nullptr, nullptr, S, Dm, Dm);

    const size_t smemBytes = (size_t)(16 * LROW + 16) * sizeof(float); // ~128.4 KB of the 320 KB LDS
    attn_kernel<<<dim3(S / 16, H), 128, smemBytes, stream>>>(Q16, K16, Vt16, bias, mask, at16);

    gemm_xWT<2><<<gGemm, 128, 0, stream>>>(at16, Wo16, nullptr, res, x, S, Dm, Dm);
    ln_kernel<<<S, 256, 0, stream>>>(res, lnw, lnb, out);
}
